// RNN_678604833045
// MI455X (gfx1250) — matrix operand order repaired and checked by reading, not compiled, not run on hardware
//
#include <hip/hip_runtime.h>
#include <hip/hip_bf16.h>

// LSTM: B=16, A=32, T=1024, N_IN=128, N_H=128
#define B_     16
#define A_     32
#define T_     1024
#define NIN    128
#define NH     128
#define ROWS   16            // (b,a) rows per block
#define NBLK   32            // 512 / ROWS
#define KDIM   256           // [h | x]
#define NGATES 512           // 4 * NH  (i,f,g,o)
#define LDWX   136           // padded LDS row stride (halfs) for x-half weights
#define LDA    264           // padded LDS row stride (halfs) for h|x staging

typedef __attribute__((ext_vector_type(16))) __bf16 v16bf;
typedef __attribute__((ext_vector_type(8)))  float  v8f;

union Frag { v16bf v; unsigned int u[8]; uint4 q[2]; };

__device__ __forceinline__ unsigned short f32_to_bf16(float f) {
  union { float f; unsigned int u; } c; c.f = f;
  unsigned int u = c.u;
  u += 0x7fffu + ((u >> 16) & 1u);   // round-to-nearest-even
  return (unsigned short)(u >> 16);
}
// pack two f32 -> {lo16: bf16(lo), hi16: bf16(hi)}
__device__ __forceinline__ unsigned int pack2_bf16(float lo, float hi) {
  union { float f; unsigned int u; } a, b; a.f = lo; b.f = hi;
  unsigned int ul = a.u + (0x7fffu + ((a.u >> 16) & 1u));
  unsigned int uh = b.u + (0x7fffu + ((b.u >> 16) & 1u));
  return (ul >> 16) | (uh & 0xffff0000u);
}

#define LOG2E   1.44269504088896340736f
#define LOG2E2  2.88539008177792681472f

__device__ __forceinline__ float sigmoidf_(float x) {
  return __builtin_amdgcn_rcpf(1.0f + __builtin_amdgcn_exp2f(-LOG2E * x));
}
__device__ __forceinline__ float tanhf_(float x) {
  // tanh(x) = 1 - 2/(e^(2x)+1); exp2 overflow -> inf -> rcp -> 0 -> +1 (safe)
  return 1.0f - 2.0f * __builtin_amdgcn_rcpf(
                           __builtin_amdgcn_exp2f(LOG2E2 * x) + 1.0f);
}

// ---------------------------------------------------------------------------
// Prologue: pack Wc[n][k] = bf16 of [Wh | Wx] (n = gate-major output row,
// k = 0..127 -> Wh col, 128..255 -> Wx col), bias[n] = bh[n] + bx[n].
// ---------------------------------------------------------------------------
__global__ void lstm_pack_weights(
    const float* __restrict__ Wii, const float* __restrict__ Wif,
    const float* __restrict__ Wig, const float* __restrict__ Wio,
    const float* __restrict__ Whi, const float* __restrict__ Whf,
    const float* __restrict__ Whg, const float* __restrict__ Who,
    const float* __restrict__ bii, const float* __restrict__ bif,
    const float* __restrict__ big, const float* __restrict__ bio,
    const float* __restrict__ bhi, const float* __restrict__ bhf,
    const float* __restrict__ bhg, const float* __restrict__ bho,
    unsigned short* __restrict__ Wc, float* __restrict__ bias)
{
  int idx = blockIdx.x * blockDim.x + threadIdx.x;    // 0 .. 512*256-1
  int n  = idx >> 8;                                  // 0..511
  int k  = idx & 255;                                 // 0..255
  int gi = n >> 7;                                    // gate 0..3 (i,f,g,o)
  int r  = n & 127;
  const float* Whp = (gi == 0) ? Whi : (gi == 1) ? Whf : (gi == 2) ? Whg : Who;
  const float* Wxp = (gi == 0) ? Wii : (gi == 1) ? Wif : (gi == 2) ? Wig : Wio;
  float v = (k < NH) ? Whp[r * NH + k] : Wxp[r * NIN + (k - NH)];
  Wc[n * KDIM + k] = f32_to_bf16(v);
  if (k == 0) {
    const float* bhp = (gi == 0) ? bhi : (gi == 1) ? bhf : (gi == 2) ? bhg : bho;
    const float* bxp = (gi == 0) ? bii : (gi == 1) ? bif : (gi == 2) ? big : bio;
    bias[n] = bhp[r] + bxp[r];
  }
}

// ---------------------------------------------------------------------------
// Persistent LSTM kernel. Each block owns ROWS=16 sequences for all T.
// Wave w owns gate columns [16w,16w+16): its accumulators are i,f,g,o for the
// SAME columns -> register-resident cell update.
// - h-weights (128 VGPRs/wave) persistent in registers.
// - x-projection runs ONE STEP AHEAD into accx (off the serial h-dependency
//   chain; its LDS-fed WMMAs overlap the elementwise VALU via co-execution).
// - X global loads pipelined two steps ahead.
// Serial chain/step: barrier -> 8 ds_load + 16 reg-B WMMA -> barrier -> cell.
// ---------------------------------------------------------------------------
__global__ void __launch_bounds__(256, 1)
lstm_persistent(const float* __restrict__ X,
                const unsigned short* __restrict__ Wc,
                const float* __restrict__ bias,
                float* __restrict__ out)
{
  extern __shared__ char smem[];
  unsigned short* sW = (unsigned short*)smem;          // 512*LDWX halfs (x-half)
  unsigned short* sA = sW + NGATES * LDWX;             // ROWS*LDA halfs

  const int tid  = threadIdx.x;
  const int lane = tid & 31;
  const int wave = tid >> 5;                 // 0..7
  const int row0 = blockIdx.x * ROWS;        // global (b*A+a) base row

  // ---- fill LDS: x-half of weights (K cols 128..255), zero h staging ----
  for (int i = tid; i < NGATES * 64; i += blockDim.x) {   // 64 uint pairs/row
    int n = i >> 6, kp = i & 63;
    ((unsigned int*)(sW + n * LDWX))[kp] =
        ((const unsigned int*)(Wc + n * KDIM + NH))[kp];
  }
  for (int i = tid; i < ROWS * NH; i += blockDim.x) {
    int r = i >> 7, k = i & 127;
    sA[r * LDA + k] = 0;                     // h_0 = 0 (bf16 zero)
  }

  // ---- per-wave column ownership + per-lane constants ----
  const int jcol = wave * 16 + (lane & 15);  // hidden column 0..127
  const int hl   = (lane >> 4);              // half-wave: 0 or 1
  const int mb   = hl << 3;                  // row block 0 or 8
  const float bi = bias[jcol];
  const float bf = bias[NH + jcol];
  const float bg = bias[2 * NH + jcol];
  const float bo = bias[3 * NH + jcol];

  // ---- persistent h-half B fragments: [gate][k0], K = k0*32 .. +31 ----
  Frag bh[4][4];
#pragma unroll
  for (int g = 0; g < 4; ++g) {
#pragma unroll
    for (int k0 = 0; k0 < 4; ++k0) {
      const uint4* wp = (const uint4*)(Wc + (size_t)(g * NH + jcol) * KDIM +
                                       k0 * 32 + (hl << 4));
      bh[g][k0].q[0] = wp[0];
      bh[g][k0].q[1] = wp[1];
    }
  }

  // x pipeline mapping: thread (er, ej) owns X cols [ej*8, ej*8+8)
  const int er = tid >> 4;                   // row 0..15
  const int ej = tid & 15;
  const float* Xrow = X + (size_t)(row0 + er) * T_ * NIN + ej * 8;
  unsigned short* sAx = sA + er * LDA + NH + ej * 8;

  // x-projection GEMM into accx (reads sA x-region, B from LDS sW)
  auto xpart = [&](v8f* axx) {
#pragma unroll
    for (int k0 = 0; k0 < 4; ++k0) {
      Frag a;
      const int kb = NH + k0 * 32 + (hl << 4);
      const uint4* ap = (const uint4*)(sA + (lane & 15) * LDA + kb);
      a.q[0] = ap[0];
      a.q[1] = ap[1];
#pragma unroll
      for (int g = 0; g < 4; ++g) {
        Frag b;
        const uint4* wp = (const uint4*)(sW + (g * NH + jcol) * LDWX +
                                         k0 * 32 + (hl << 4));
        b.q[0] = wp[0];
        b.q[1] = wp[1];
        axx[g] = (k0 == 0)
            ? __builtin_amdgcn_wmma_f32_16x16x32_bf16(
                  false, a.v, false, b.v, (short)0, (v8f)0.0f, false, false)
            : __builtin_amdgcn_wmma_f32_16x16x32_bf16(
                  false, a.v, false, b.v, (short)0, axx[g], false, false);
      }
    }
  };
  auto stage_x = [&](const float4& xa, const float4& xb) {
    uint4 p;
    p.x = pack2_bf16(xa.x, xa.y);
    p.y = pack2_bf16(xa.z, xa.w);
    p.z = pack2_bf16(xb.x, xb.y);
    p.w = pack2_bf16(xb.z, xb.w);
    *(uint4*)sAx = p;
  };

  // ---- prologue: stage x_0, accx = Wx*x_0, preload x_1 ----
  float4 xa, xb;
  {
    const float4* xp = (const float4*)(Xrow);          // x_0
    xa = xp[0]; xb = xp[1];
  }
  stage_x(xa, xb);
  __syncthreads();                 // sW + sA(x_0, h=0) visible

  v8f accx[4];
  xpart(accx);                     // accx = Wx * x_0

  {
    const float4* xp = (const float4*)(Xrow + NIN);    // x_1
    xa = xp[0]; xb = xp[1];
  }
  __syncthreads();                 // prologue x-region reads done

  // cell state: c for (row mb+r, col jcol), r = 0..7
  float cst[8];
#pragma unroll
  for (int r = 0; r < 8; ++r) cst[r] = 0.0f;

  for (int t = 0; t < T_; ++t) {
    // ---- stage x_{t+1} (regs loaded one iter earlier) ----
    if (t + 1 < T_) stage_x(xa, xb);
    __syncthreads();    // h_{t-1} + x_{t+1} staged, prev reads done

    // ---- issue X loads for t+2 (hidden behind the GEMMs) ----
    if (t + 2 < T_) {
      const float4* xp = (const float4*)(Xrow + (size_t)(t + 2) * NIN);
      xa = xp[0]; xb = xp[1];
    }

    // ---- serial part: gates_t = accx + Wh*h_{t-1} (all-register B) ----
    v8f acc[4];
#pragma unroll
    for (int g = 0; g < 4; ++g) acc[g] = accx[g];
#pragma unroll
    for (int k0 = 0; k0 < 4; ++k0) {
      Frag a;
      const int kb = k0 * 32 + (hl << 4);
      const uint4* ap = (const uint4*)(sA + (lane & 15) * LDA + kb);
      a.q[0] = ap[0];          // K = kb .. kb+7
      a.q[1] = ap[1];          // K = kb+8 .. kb+15
#pragma unroll
      for (int g = 0; g < 4; ++g) {
        acc[g] = __builtin_amdgcn_wmma_f32_16x16x32_bf16(
            false, a.v, false, bh[g][k0].v, (short)0, acc[g], false, false);
      }
    }

    // ---- look-ahead: accx = Wx * x_{t+1} (off the dependency chain) ----
    xpart(accx);

    __syncthreads();    // all LDS reads of sA complete -> safe to rewrite h

    // ---- register-resident LSTM cell update (co-executes w/ x-part tail) --
#pragma unroll
    for (int r = 0; r < 8; ++r) {
      float ig = sigmoidf_(acc[0][r] + bi);
      float fg = sigmoidf_(acc[1][r] + bf);
      float gg = tanhf_(acc[2][r] + bg);
      float og = sigmoidf_(acc[3][r] + bo);
      cst[r] = fg * cst[r] + ig * gg;
      float h = og * tanhf_(cst[r]);
      out[((size_t)(row0 + mb + r) * T_ + t) * NH + jcol] = h;
      sA[(mb + r) * LDA + jcol] = f32_to_bf16(h);   // h_t for next step
    }
  }
}

// ---------------------------------------------------------------------------
extern "C" void kernel_launch(void* const* d_in, const int* in_sizes, int n_in,
                              void* d_out, int out_size, void* d_ws, size_t ws_size,
                              hipStream_t stream) {
  (void)in_sizes; (void)n_in; (void)out_size; (void)ws_size;
  const float* X = (const float*)d_in[0];
  // dict order: X, (W_ii,b_ii,W_if,b_if,W_ig,b_ig,W_io,b_io),
  //             (W_hi,b_hi,W_hf,b_hf,W_hg,b_hg,W_ho,b_ho)
  unsigned short* Wc = (unsigned short*)d_ws;                       // 512*256 bf16
  float* bias = (float*)((char*)d_ws + (size_t)NGATES * KDIM * 2);  // 512 f32

  lstm_pack_weights<<<512, 256, 0, stream>>>(
      (const float*)d_in[1],  (const float*)d_in[3],
      (const float*)d_in[5],  (const float*)d_in[7],
      (const float*)d_in[9],  (const float*)d_in[11],
      (const float*)d_in[13], (const float*)d_in[15],
      (const float*)d_in[2],  (const float*)d_in[4],
      (const float*)d_in[6],  (const float*)d_in[8],
      (const float*)d_in[10], (const float*)d_in[12],
      (const float*)d_in[14], (const float*)d_in[16],
      Wc, bias);

  constexpr size_t smem_bytes =
      (size_t)(NGATES * LDWX + ROWS * LDA) * 2;     // x-half weights + staging
  static_assert(smem_bytes <= 320u * 1024u, "exceeds 320KB WGP LDS");
  hipFuncSetAttribute((const void*)lstm_persistent,
                      hipFuncAttributeMaxDynamicSharedMemorySize,
                      (int)smem_bytes);
  lstm_persistent<<<NBLK, 256, smem_bytes, stream>>>(X, Wc, bias, (float*)d_out);
}
